// MultiHeadAttention_26834955665743
// MI455X (gfx1250) — compile-verified, matrix operand order checked by reading, not run on hardware
//
#include <hip/hip_runtime.h>

// ---------------------------------------------------------------------------
// MHA forward for MI455X (gfx1250, wave32, WMMA).
//  B=4, S=2048, D=1024, H=16, HD=64.  ~137 GFLOP, ~190 MB traffic -> balanced
//  at bf16 WMMA rate; all matmuls go through v_wmma_f32_16x16x32_bf16 and
//  tile staging goes through GLOBAL_LOAD_ASYNC_TO_LDS (ASYNCcnt) when the
//  toolchain exposes the builtin.
// ---------------------------------------------------------------------------

typedef __bf16 bf16;
typedef __attribute__((ext_vector_type(4)))  bf16  v4bf;
typedef __attribute__((ext_vector_type(8)))  bf16  v8bf;
typedef __attribute__((ext_vector_type(16))) bf16  v16bf;
typedef __attribute__((ext_vector_type(8)))  float v8f;
typedef __attribute__((__vector_size__(4 * sizeof(int)))) int v4i;  // builtin arg type

#if defined(__has_builtin)
#if __has_builtin(__builtin_amdgcn_global_load_async_to_lds_b128) && \
    __has_builtin(__builtin_amdgcn_s_wait_asynccnt)
#define USE_ASYNC_LDS 1
#endif
#endif
#ifndef USE_ASYNC_LDS
#define USE_ASYNC_LDS 0
#endif

// 16-byte global -> LDS copy; async (ASYNCcnt) when available.
static __device__ __forceinline__ void cp16_g2l(const bf16* g, bf16* l) {
#if USE_ASYNC_LDS
  __builtin_amdgcn_global_load_async_to_lds_b128((v4i*)g, (v4i*)l, 0, 0);
#else
  *(v8bf*)l = *(const v8bf*)g;
#endif
}
static __device__ __forceinline__ void cp_wait() {
#if USE_ASYNC_LDS
  __builtin_amdgcn_s_wait_asynccnt(0);
#endif
}

// A/B fragment load for a row-major [16 x 32] bf16 tile (row stride in elems).
// Per ISA 7.12.2: lane l holds row (l&15); its 16 elements are
// k = 8*(l>>4)+0..7 then 16+8*(l>>4)+0..7 -> two contiguous 16B chunks.
static __device__ __forceinline__ v16bf frag_ld(const bf16* row, int lh) {
  v8bf lo = *(const v8bf*)(row + 8 * lh);
  v8bf hi = *(const v8bf*)(row + 16 + 8 * lh);
  return __builtin_shufflevector(lo, hi, 0, 1, 2, 3, 4, 5, 6, 7,
                                 8, 9, 10, 11, 12, 13, 14, 15);
}

static __device__ __forceinline__ v8f wmma_bf16(v16bf a, v16bf b, v8f c) {
  return __builtin_amdgcn_wmma_f32_16x16x32_bf16(false, a, false, b,
                                                 (short)0, c, false, false);
}

// ---------------------------------------------------------------------------
// fp32 -> bf16 conversion (vectorized x4)
// ---------------------------------------------------------------------------
__global__ __launch_bounds__(256) void cvt_f32_bf16_k(
    const float* __restrict__ in, bf16* __restrict__ out, int n4) {
  int i = blockIdx.x * 256 + threadIdx.x;
  if (i < n4) {
    float4 v = ((const float4*)in)[i];
    v4bf o;
    o.x = (bf16)v.x; o.y = (bf16)v.y; o.z = (bf16)v.z; o.w = (bf16)v.w;
    ((v4bf*)out)[i] = o;
  }
}

// ---------------------------------------------------------------------------
// C[M,N] = A[M,K] * W[N,K]^T + bias[N]   (bf16 in, fp32 accum)
// 256 threads = 8 waves; block tile 128(M) x 128(N); k-step 32; LDS double-
// buffered with async staging. Wave (wm,wn) computes 32x64 = 2x4 WMMA tiles.
// ---------------------------------------------------------------------------
template <bool OUT_F32>
__global__ __launch_bounds__(256) void gemm_bf16_wmma_k(
    const bf16* __restrict__ A, const bf16* __restrict__ W,
    const float* __restrict__ bias, bf16* __restrict__ Cb,
    float* __restrict__ Cf, int M, int N, int K) {
  __shared__ bf16 As[2][128 * 32];  // 2 x 8 KB
  __shared__ bf16 Bs[2][128 * 32];  // 2 x 8 KB

  const int tid = threadIdx.x;
  const int lane = tid & 31;
  const int lh = lane >> 4;
  const int wid = tid >> 5;
  const int wm = wid & 3;   // M offset wm*32
  const int wn = wid >> 2;  // N offset wn*64
  const long rowBase = (long)blockIdx.y * 128;
  const long colBase = (long)blockIdx.x * 128;

  // Stage one 128x32 A tile + 128x32 B(W) tile: 1024 x 16B chunks, 4/thread.
  auto stage = [&](int buf, int k0) {
#pragma unroll
    for (int i = 0; i < 2; ++i) {
      int ch = tid + i * 256;
      int r = ch >> 2, ko = (ch & 3) * 8;
      cp16_g2l(A + (rowBase + r) * K + k0 + ko, &As[buf][r * 32 + ko]);
      cp16_g2l(W + (colBase + r) * K + k0 + ko, &Bs[buf][r * 32 + ko]);
    }
  };

  v8f acc[2][4] = {};
  stage(0, 0);
  int buf = 0;
  for (int k0 = 0; k0 < K; k0 += 32) {
    cp_wait();
    __syncthreads();
    if (k0 + 32 < K) stage(buf ^ 1, k0 + 32);  // overlap with WMMAs below

    v16bf a0 = frag_ld(&As[buf][(wm * 32 + (lane & 15)) * 32], lh);
    v16bf a1 = frag_ld(&As[buf][(wm * 32 + 16 + (lane & 15)) * 32], lh);
#pragma unroll
    for (int tj = 0; tj < 4; ++tj) {
      v16bf bfr = frag_ld(&Bs[buf][(wn * 64 + tj * 16 + (lane & 15)) * 32], lh);
      acc[0][tj] = wmma_bf16(a0, bfr, acc[0][tj]);
      acc[1][tj] = wmma_bf16(a1, bfr, acc[1][tj]);
    }
    buf ^= 1;
  }

  // Epilogue: C layout = lane holds col (lane&15), rows j + 8*(lane>>4).
#pragma unroll
  for (int ti = 0; ti < 2; ++ti)
#pragma unroll
    for (int tj = 0; tj < 4; ++tj) {
      long col = colBase + wn * 64 + tj * 16 + (lane & 15);
      float bb = bias[col];
#pragma unroll
      for (int j = 0; j < 8; ++j) {
        long row = rowBase + wm * 32 + ti * 16 + j + 8 * lh;
        float v = acc[ti][tj][j] + bb;
        if (OUT_F32) Cf[row * N + col] = v;
        else         Cb[row * N + col] = (bf16)v;
      }
    }
}

// ---------------------------------------------------------------------------
// Flash attention over the "faithful reshape" layout.
// QKV is bf16 [B, 2048, 3072]; head h, position s, channel c lives at
//   row = h*128 + (s>>4), col = (s&15)*192 + c      (c: Q=0..63 K=64.. V=128..)
// Grid (S/64, H, B); 128 threads = 4 waves; wave w handles 16 query rows.
// Per 32-key chunk: K staged to LDS via async copy, V staged transposed,
// QK^T via 4 WMMAs, online softmax, P through LDS to A-layout, P*V 4 WMMAs.
// ---------------------------------------------------------------------------
__global__ __launch_bounds__(128) void flash_attn_wmma_k(
    const bf16* __restrict__ QKV, bf16* __restrict__ vals) {
  __shared__ bf16 Kc[32 * 64];     // K chunk row-major [key][hd], 4 KB
  __shared__ bf16 Vt[64 * 32];     // V^T chunk [hd][key], 4 KB
  __shared__ bf16 Pt[4][16 * 32];  // per-wave P tile, 4 KB

  const int tid = threadIdx.x;
  const int lane = tid & 31, lh = lane >> 4, w = tid >> 5;
  const int h = blockIdx.y, b = blockIdx.z;
  const bf16* Qb = QKV + (long)b * 2048 * 3072;

  auto rowptr = [&](int s, int c) -> const bf16* {
    return Qb + (long)(h * 128 + (s >> 4)) * 3072 + (s & 15) * 192 + c;
  };

  // Q fragments for this lane's row, hd 0..31 and 32..63 (loaded once).
  const int sq = blockIdx.x * 64 + w * 16 + (lane & 15);
  const bf16* qrow = rowptr(sq, 0);
  v16bf qa0 = frag_ld(qrow, lh);
  v16bf qa1 = frag_ld(qrow + 32, lh);

  float m[8], lsum[8];
  v8f o[4] = {};
#pragma unroll
  for (int j = 0; j < 8; ++j) { m[j] = -__builtin_inff(); lsum[j] = 0.f; }

  for (int c = 0; c < 2048; c += 32) {
    __syncthreads();
    // Stage K chunk (32 keys x 64 hd): 256 x 16B chunks, 2/thread, async.
    // Stage V^T: same source shape, but scattered to [hd][key].
#pragma unroll
    for (int i = 0; i < 2; ++i) {
      int ch = tid + i * 128;
      int key = ch >> 3, ho = (ch & 7) * 8;
      cp16_g2l(rowptr(c + key, 64 + ho), Kc + key * 64 + ho);
      v8bf vv = *(const v8bf*)(rowptr(c + key, 128 + ho));
#pragma unroll
      for (int e = 0; e < 8; ++e) Vt[(ho + e) * 32 + key] = vv[e];
    }
    cp_wait();
    __syncthreads();

    // Scores for 32 keys from LDS K: two 16x16 C tiles.
    v8f s0 = {}, s1 = {};
    {
      const bf16* kr0 = Kc + (lane & 15) * 64;
      s0 = wmma_bf16(qa0, frag_ld(kr0, lh), s0);
      s0 = wmma_bf16(qa1, frag_ld(kr0 + 32, lh), s0);
      const bf16* kr1 = Kc + (16 + (lane & 15)) * 64;
      s1 = wmma_bf16(qa0, frag_ld(kr1, lh), s1);
      s1 = wmma_bf16(qa1, frag_ld(kr1 + 32, lh), s1);
    }

    // Online softmax (scale 0.125). Row r = j + 8*lh; reduce across the 16
    // columns (lanes sharing lane>>4) with shfl_xor.
#pragma unroll
    for (int j = 0; j < 8; ++j) {
      float x0 = s0[j] * 0.125f, x1 = s1[j] * 0.125f;
      float rmax = fmaxf(x0, x1);
      for (int d = 1; d < 16; d <<= 1) rmax = fmaxf(rmax, __shfl_xor(rmax, d, 32));
      float mn = fmaxf(m[j], rmax);
      float alpha = __expf(m[j] - mn);
      float p0 = __expf(x0 - mn), p1 = __expf(x1 - mn);
      m[j] = mn;
      float rs = p0 + p1;
      for (int d = 1; d < 16; d <<= 1) rs += __shfl_xor(rs, d, 32);
      lsum[j] = lsum[j] * alpha + rs;
#pragma unroll
      for (int nb = 0; nb < 4; ++nb) o[nb][j] *= alpha;
      int r = j + 8 * lh;
      Pt[w][r * 32 + (lane & 15)]      = (bf16)p0;   // keys c..c+15
      Pt[w][r * 32 + 16 + (lane & 15)] = (bf16)p1;   // keys c+16..c+31
    }

    // P (16x32, A-layout via LDS) x V^T fragments -> accumulate O (16x64).
    v16bf pa = frag_ld(&Pt[w][(lane & 15) * 32], lh);
#pragma unroll
    for (int nb = 0; nb < 4; ++nb)
      o[nb] = wmma_bf16(pa, frag_ld(Vt + (nb * 16 + (lane & 15)) * 32, lh), o[nb]);
  }

  // Normalize and store straight into the reshaped [B,S,D] vals layout:
  // per batch, linear offset h*(2048*64) + s*64 + hd == row-major [2048,1024].
#pragma unroll
  for (int nb = 0; nb < 4; ++nb) {
    int hd = nb * 16 + (lane & 15);
#pragma unroll
    for (int j = 0; j < 8; ++j) {
      int r = j + 8 * lh;
      int s2 = blockIdx.x * 64 + w * 16 + r;
      float v = o[nb][j] / lsum[j];
      vals[(long)b * 2097152 + (long)h * 131072 + (long)s2 * 64 + hd] = (bf16)v;
    }
  }
}

// ---------------------------------------------------------------------------
// Launch: cvt(X,W_in,W_out) -> GEMM(QKV) -> flash-attn -> GEMM(out proj)
// Workspace layout (bytes, 16B aligned):
//   Xb   @ 0         : 16 MB   Wb  @ 16777216 : 6 MB   Wob @ 23068672 : 2 MB
//   QKVb @ 25165824  : 48 MB   Vals@ 75497472 : 16 MB  (total ~88 MB)
// ---------------------------------------------------------------------------
extern "C" void kernel_launch(void* const* d_in, const int* in_sizes, int n_in,
                              void* d_out, int out_size, void* d_ws,
                              size_t ws_size, hipStream_t stream) {
  const float* X     = (const float*)d_in[0];
  const float* W_in  = (const float*)d_in[1];
  const float* b_in  = (const float*)d_in[2];
  const float* W_out = (const float*)d_in[3];
  const float* b_out = (const float*)d_in[4];

  char* ws = (char*)d_ws;
  bf16* Xb   = (bf16*)(ws + 0);
  bf16* Wb   = (bf16*)(ws + 16777216);
  bf16* Wob  = (bf16*)(ws + 23068672);
  bf16* QKVb = (bf16*)(ws + 25165824);
  bf16* Vals = (bf16*)(ws + 75497472);

  cvt_f32_bf16_k<<<8192, 256, 0, stream>>>(X, Xb, 2097152);
  cvt_f32_bf16_k<<<3072, 256, 0, stream>>>(W_in, Wb, 786432);
  cvt_f32_bf16_k<<<1024, 256, 0, stream>>>(W_out, Wob, 262144);

  // QKV = X * W_in^T + b_in : M=8192, N=3072, K=1024, bf16 out.
  gemm_bf16_wmma_k<false><<<dim3(3072 / 128, 8192 / 128), 256, 0, stream>>>(
      Xb, Wb, b_in, QKVb, nullptr, 8192, 3072, 1024);

  // Attention: grid (S/64, H, B).
  flash_attn_wmma_k<<<dim3(32, 16, 4), 128, 0, stream>>>(QKVb, Vals);

  // out = vals * W_out^T + b_out : M=8192, N=1024, K=1024, fp32 out.
  gemm_bf16_wmma_k<true><<<dim3(1024 / 128, 8192 / 128), 256, 0, stream>>>(
      Vals, Wob, b_out, nullptr, (float*)d_out, 8192, 1024, 1024);
}
